// InteractionModule_90254442758879
// MI455X (gfx1250) — compile-verified, matrix operand order checked by reading, not run on hardware
//
#include <hip/hip_runtime.h>

// PhysNet InteractionModule fused for gfx1250 (wave32, WMMA f16->f32).
// N=50000 nodes, F=128 feats, K=64 edge feats, E=800000 edges, 3 residual layers.
//
// Pipeline: [cvt weights->f16] -> [zero aggr] -> [node GEMMs] -> [edge gate
// GEMM + gather/scatter] -> [residual chain + final]. h (f16, 12.8MB) and
// aggr (25.6MB) are L2-resident on the 192MB L2, so the random edge phase
// stays on-chip; HBM traffic ~300MB compulsory.

#define LN2F   0.69314718055994530942f
#define LOG2EF 1.44269504088896340736f

typedef __attribute__((ext_vector_type(16))) _Float16 v16h;
typedef __attribute__((ext_vector_type(8)))  float    v8f;

// shifted softplus via v_exp_f32/v_log_f32 (2^x / log2 x) only:
// ssp(t) = max(t,0) + (log2(1 + 2^(-|t|*log2e)) - 1) * ln2
static __device__ __forceinline__ float sspf(float t) {
  float a = __builtin_fabsf(t);
  float e = __builtin_amdgcn_exp2f(-a * LOG2EF);  // v_exp_f32: 2^x = e^{-|t|}
  float l = __builtin_amdgcn_logf(1.0f + e);      // v_log_f32: log2(1+e^{-|t|})
  return fmaxf(t, 0.0f) + (l - 1.0f) * LN2F;
}

union H16 { v16h v; _Float16 h[16]; };

// 16x32 f16 A/B fragment per CDNA5 ISA 7.12.2 (two contiguous 8-half runs ->
// two ds_load_b128 per fragment):
// lane<16 : row r0+lane,    k in {kb..kb+7} U {kb+16..kb+23}
// lane>=16: row r0+lane-16, k in {kb+8..kb+15} U {kb+24..kb+31}
static __device__ __forceinline__ v16h load_frag(const _Float16* base, int ldm,
                                                 int r0, int kbase, int lane) {
  const _Float16* p = base + (r0 + (lane & 15)) * ldm + kbase + ((lane >> 4) << 3);
  H16 u;
#pragma unroll
  for (int e = 0; e < 8; ++e) { u.h[e] = p[e]; u.h[e + 8] = p[e + 16]; }
  return u.v;
}

static __device__ __forceinline__ v8f wmma_f16(v16h a, v16h b, v8f c) {
  return __builtin_amdgcn_wmma_f32_16x16x32_f16(false, a, false, b,
                                                (short)0, c, false, false);
}

// stage pre-converted f16 weights global->LDS as b128 copies
static __device__ __forceinline__ void lds_copy(_Float16* dst, const _Float16* src,
                                                int bytes, int tid) {
  const float4* s = (const float4*)src;
  float4* d = (float4*)dst;
  int n = bytes >> 4;
  for (int i = tid; i < n; i += 256) d[i] = s[i];
}

// ---------------------------------------------------------------------------
// K_cvt: f32 -> f16 weight conversion (run once per launch, tiny)
// ---------------------------------------------------------------------------
__global__ __launch_bounds__(256) void cvt_kernel(const float* __restrict__ src,
                                                  _Float16* __restrict__ dst, int n4) {
  int i = blockIdx.x * 256 + threadIdx.x;
  if (i < n4) {
    float4 v = *(const float4*)(src + (long)i * 4);
    dst[(long)i * 4 + 0] = (_Float16)v.x;
    dst[(long)i * 4 + 1] = (_Float16)v.y;
    dst[(long)i * 4 + 2] = (_Float16)v.z;
    dst[(long)i * 4 + 3] = (_Float16)v.w;
  }
}

// ---------------------------------------------------------------------------
// K0: zero the aggregation buffer (graph-capture safe)
// ---------------------------------------------------------------------------
__global__ __launch_bounds__(256) void zero_kernel(float* __restrict__ p, long n4) {
  long i = (long)blockIdx.x * 256 + threadIdx.x;
  if (i < n4) *(float4*)(p + i * 4) = make_float4(0.f, 0.f, 0.f, 0.f);
}

// ---------------------------------------------------------------------------
// K1: xa = ssp(x); h = ssp(xa@W_diff^T + b_diff) -> f16 ws buffer
//                  s = ssp(xa@W_same^T + b_same) -> msged_x slot of d_out
// 128 rows/block, 8 waves, each wave owns one 16-row stripe (8 WMMA tiles).
// ---------------------------------------------------------------------------
__global__ __launch_bounds__(256) void node_linear_kernel(
    const float* __restrict__ x,
    const _Float16* __restrict__ wDiff16, const float* __restrict__ b_diff,
    const _Float16* __restrict__ wSame16, const float* __restrict__ b_same,
    _Float16* __restrict__ hbuf, float* __restrict__ sOut, int N) {
  __shared__ _Float16 sA[128 * 128];   // 32 KB activated x tile
  extern __shared__ _Float16 sW[];     // 32 KB dynamic weight buffer
  const int tid = threadIdx.x, lane = tid & 31, wave = tid >> 5;
  const int row0 = blockIdx.x * 128;
  const bool full = (row0 + 128) <= N;

  for (int i = tid; i < 128 * 32; i += 256) {
    int r = i >> 5, c4 = (i & 31) << 2;
    int row = row0 + r;
    float4 v = make_float4(0.f, 0.f, 0.f, 0.f);
    if (full || row < N) v = *(const float4*)(x + (long)row * 128 + c4);
    sA[r * 128 + c4 + 0] = (_Float16)sspf(v.x);
    sA[r * 128 + c4 + 1] = (_Float16)sspf(v.y);
    sA[r * 128 + c4 + 2] = (_Float16)sspf(v.z);
    sA[r * 128 + c4 + 3] = (_Float16)sspf(v.w);
  }

  const int mt = wave;  // 0..7
  for (int phase = 0; phase < 2; ++phase) {
    const _Float16* W16 = phase ? wSame16 : wDiff16;
    const float* b = phase ? b_same : b_diff;
    __syncthreads();  // sA ready / previous phase done with sW
    lds_copy(sW, W16, 128 * 128 * 2, tid);
    if (phase == 0)
      __builtin_prefetch((const char*)wSame16 + tid * 128, 0, 1);
    __syncthreads();

    for (int ct = 0; ct < 8; ++ct) {
      v8f acc = {};
#pragma unroll
      for (int kc = 0; kc < 4; ++kc)
        acc = wmma_f16(load_frag(sA, 128, mt * 16, kc * 32, lane),
                       load_frag(sW, 128, ct * 16, kc * 32, lane), acc);
      int n = ct * 16 + (lane & 15);
      float bias = b[n];
      int rb = row0 + mt * 16 + ((lane >> 4) << 3);
      if (phase == 0) {
        if (full) {
#pragma unroll
          for (int r = 0; r < 8; ++r)
            hbuf[(long)(rb + r) * 128 + n] = (_Float16)sspf(acc[r] + bias);
        } else {
#pragma unroll
          for (int r = 0; r < 8; ++r)
            if (rb + r < N) hbuf[(long)(rb + r) * 128 + n] = (_Float16)sspf(acc[r] + bias);
        }
      } else {
        if (full) {
#pragma unroll
          for (int r = 0; r < 8; ++r)
            sOut[(long)(rb + r) * 128 + n] = sspf(acc[r] + bias);
        } else {
#pragma unroll
          for (int r = 0; r < 8; ++r)
            if (rb + r < N) sOut[(long)(rb + r) * 128 + n] = sspf(acc[r] + bias);
        }
      }
    }
  }
}

// ---------------------------------------------------------------------------
// K2: gate = edge_attr@W_G^T (WMMA); msg = h[src]*gate; atomic scatter to aggr
// 128 edges/block; h is f16 + L2-resident -> cheap random gather.
// ---------------------------------------------------------------------------
__global__ __launch_bounds__(256) void edge_kernel(
    const float* __restrict__ edge_attr, const long long* __restrict__ edge_index,
    const _Float16* __restrict__ wG16, const _Float16* __restrict__ hbuf,
    float* __restrict__ aggr, int E) {
  __shared__ _Float16 sEA[128 * 64];  // 16 KB
  __shared__ _Float16 sWG[128 * 64];  // 16 KB
  __shared__ int sSrc[128];
  __shared__ int sDst[128];
  const int tid = threadIdx.x, lane = tid & 31, wave = tid >> 5;
  const long e0 = (long)blockIdx.x * 128;  // E % 128 == 0

  for (int i = tid; i < 128 * 16; i += 256) {
    int r = i >> 4, c4 = (i & 15) << 2;
    float4 v = *(const float4*)(edge_attr + (e0 + r) * 64 + c4);
    sEA[r * 64 + c4 + 0] = (_Float16)v.x;
    sEA[r * 64 + c4 + 1] = (_Float16)v.y;
    sEA[r * 64 + c4 + 2] = (_Float16)v.z;
    sEA[r * 64 + c4 + 3] = (_Float16)v.w;
  }
  lds_copy(sWG, wG16, 128 * 64 * 2, tid);
  if (tid < 128) {
    sSrc[tid] = (int)edge_index[e0 + tid];            // row 0: src
    sDst[tid] = (int)edge_index[(long)E + e0 + tid];  // row 1: dst
  }
  __syncthreads();

  const int mt = wave;  // 0..7
  for (int ct = 0; ct < 8; ++ct) {
    v8f acc = {};
#pragma unroll
    for (int kc = 0; kc < 2; ++kc)
      acc = wmma_f16(load_frag(sEA, 64, mt * 16, kc * 32, lane),
                     load_frag(sWG, 64, ct * 16, kc * 32, lane), acc);
    int n = ct * 16 + (lane & 15);
    int mb = mt * 16 + ((lane >> 4) << 3);
#pragma unroll
    for (int r = 0; r < 8; ++r) {
      int src = sSrc[mb + r], dst = sDst[mb + r];
      float msg = acc[r] * (float)hbuf[(long)src * 128 + n];
      atomicAdd(&aggr[(long)dst * 128 + n], msg);  // global_atomic_add_f32 (L2)
    }
  }
}

// ---------------------------------------------------------------------------
// K3: t = s + aggr (emit msged_x); 3x residual {t += W2^T.ssp(W1^T.ssp(t)+b1)+b2};
//     out0 = ssp(t)@W_last^T + b_last + x*u. 64 rows/block, t lives in LDS f32.
// ---------------------------------------------------------------------------
__global__ __launch_bounds__(256) void residual_kernel(
    const float* __restrict__ x, const float* __restrict__ aggr,
    const _Float16* __restrict__ wRes1, const float* __restrict__ res_b1,
    const _Float16* __restrict__ wRes2, const float* __restrict__ res_b2,
    const _Float16* __restrict__ wLast16, const float* __restrict__ b_last,
    const float* __restrict__ u,
    float* __restrict__ out0, float* __restrict__ outMsg, int N) {
  __shared__ float    sT[64 * 128];    // 32 KB residual state
  __shared__ _Float16 sAct[64 * 128];  // 16 KB activated operand
  extern __shared__ _Float16 sW[];     // 32 KB dynamic weight buffer
  const int tid = threadIdx.x, lane = tid & 31, wave = tid >> 5;
  const int row0 = blockIdx.x * 64;
  const bool full = (row0 + 64) <= N;

  for (int i = tid; i < 64 * 32; i += 256) {
    int r = i >> 5, c4 = (i & 31) << 2;
    int row = row0 + r;
    float4 t = make_float4(0.f, 0.f, 0.f, 0.f);
    if (full || row < N) {
      float4 s = *(const float4*)(outMsg + (long)row * 128 + c4);
      float4 a = *(const float4*)(aggr + (long)row * 128 + c4);
      t = make_float4(s.x + a.x, s.y + a.y, s.z + a.z, s.w + a.w);
      *(float4*)(outMsg + (long)row * 128 + c4) = t;  // msged_x output
    }
    *(float4*)(sT + r * 128 + c4) = t;
  }

  const int mt = wave >> 1;           // 0..3
  const int ctBase = (wave & 1) * 4;  // 4 col tiles per wave

  for (int layer = 0; layer < 3; ++layer) {
    __syncthreads();  // sT complete, sW/sAct free
    for (int i = tid; i < 64 * 128; i += 256) sAct[i] = (_Float16)sspf(sT[i]);
    lds_copy(sW, wRes1 + layer * 128 * 128, 128 * 128 * 2, tid);
    __builtin_prefetch((const char*)(wRes2 + layer * 128 * 128) + tid * 128, 0, 1);
    __syncthreads();

    float h1v[4][8];
    for (int j = 0; j < 4; ++j) {
      int ct = ctBase + j;
      v8f acc = {};
#pragma unroll
      for (int kc = 0; kc < 4; ++kc)
        acc = wmma_f16(load_frag(sAct, 128, mt * 16, kc * 32, lane),
                       load_frag(sW, 128, ct * 16, kc * 32, lane), acc);
      float bias = res_b1[layer * 128 + ct * 16 + (lane & 15)];
#pragma unroll
      for (int r = 0; r < 8; ++r) h1v[j][r] = sspf(acc[r] + bias);
    }
    __syncthreads();  // done reading sAct / sW

    for (int j = 0; j < 4; ++j) {
      int ct = ctBase + j;
      int n = ct * 16 + (lane & 15);
      int mb = mt * 16 + ((lane >> 4) << 3);
#pragma unroll
      for (int r = 0; r < 8; ++r) sAct[(mb + r) * 128 + n] = (_Float16)h1v[j][r];
    }
    lds_copy(sW, wRes2 + layer * 128 * 128, 128 * 128 * 2, tid);
    const _Float16* nxt = (layer < 2) ? (wRes1 + (layer + 1) * 128 * 128) : wLast16;
    __builtin_prefetch((const char*)nxt + tid * 128, 0, 1);
    __syncthreads();

    for (int j = 0; j < 4; ++j) {
      int ct = ctBase + j;
      v8f acc = {};
#pragma unroll
      for (int kc = 0; kc < 4; ++kc)
        acc = wmma_f16(load_frag(sAct, 128, mt * 16, kc * 32, lane),
                       load_frag(sW, 128, ct * 16, kc * 32, lane), acc);
      int n = ct * 16 + (lane & 15);
      float bias = res_b2[layer * 128 + n];
      int mb = mt * 16 + ((lane >> 4) << 3);
#pragma unroll
      for (int r = 0; r < 8; ++r) sT[(mb + r) * 128 + n] += acc[r] + bias;
    }
  }

  // final: v = ssp(t)@W_last^T + b_last; out0 = v + x*u
  __syncthreads();
  for (int i = tid; i < 64 * 128; i += 256) sAct[i] = (_Float16)sspf(sT[i]);
  lds_copy(sW, wLast16, 128 * 128 * 2, tid);
  __syncthreads();
  for (int j = 0; j < 4; ++j) {
    int ct = ctBase + j;
    v8f acc = {};
#pragma unroll
    for (int kc = 0; kc < 4; ++kc)
      acc = wmma_f16(load_frag(sAct, 128, mt * 16, kc * 32, lane),
                     load_frag(sW, 128, ct * 16, kc * 32, lane), acc);
    int n = ct * 16 + (lane & 15);
    float bias = b_last[n], uv = u[n];
    int rb = row0 + mt * 16 + ((lane >> 4) << 3);
    if (full) {
#pragma unroll
      for (int r = 0; r < 8; ++r)
        out0[(long)(rb + r) * 128 + n] =
            acc[r] + bias + x[(long)(rb + r) * 128 + n] * uv;
    } else {
#pragma unroll
      for (int r = 0; r < 8; ++r)
        if (rb + r < N)
          out0[(long)(rb + r) * 128 + n] =
              acc[r] + bias + x[(long)(rb + r) * 128 + n] * uv;
    }
  }
}

// ---------------------------------------------------------------------------
extern "C" void kernel_launch(void* const* d_in, const int* in_sizes, int n_in,
                              void* d_out, int out_size, void* d_ws, size_t ws_size,
                              hipStream_t stream) {
  constexpr int N = 50000, F = 128, E = 800000;
  constexpr long NF = (long)N * F;
  const float*     x          = (const float*)d_in[0];
  const long long* edge_index = (const long long*)d_in[1];
  const float*     edge_attr  = (const float*)d_in[2];
  const float*     W_same     = (const float*)d_in[3];
  const float*     b_same     = (const float*)d_in[4];
  const float*     W_diff     = (const float*)d_in[5];
  const float*     b_diff     = (const float*)d_in[6];
  const float*     W_G        = (const float*)d_in[7];
  const float*     res_W1     = (const float*)d_in[8];
  const float*     res_b1     = (const float*)d_in[9];
  const float*     res_W2     = (const float*)d_in[10];
  const float*     res_b2     = (const float*)d_in[11];
  const float*     W_last     = (const float*)d_in[12];
  const float*     b_last     = (const float*)d_in[13];
  const float*     u          = (const float*)d_in[14];

  float* out0   = (float*)d_out;        // v + x*u
  float* outMsg = out0 + NF;            // msged_x (holds s between K1 and K3)

  // workspace: h (f16, NF) | aggr (f32, NF) | f16 weights (~304 KB)
  _Float16* hbuf = (_Float16*)d_ws;
  float*    aggr = (float*)((char*)d_ws + NF * 2);
  _Float16* w16  = (_Float16*)((char*)d_ws + NF * 2 + NF * 4);
  _Float16* wDiff16 = w16;                 // 16384
  _Float16* wSame16 = w16 + 16384;         // 16384
  _Float16* wG16    = w16 + 32768;         // 8192
  _Float16* wRes1   = w16 + 40960;         // 3*16384
  _Float16* wRes2   = w16 + 90112;         // 3*16384
  _Float16* wLast16 = w16 + 139264;        // 16384

  cvt_kernel<<<16, 256, 0, stream>>>(W_diff, wDiff16, 4096);
  cvt_kernel<<<16, 256, 0, stream>>>(W_same, wSame16, 4096);
  cvt_kernel<<< 8, 256, 0, stream>>>(W_G,    wG16,    2048);
  cvt_kernel<<<48, 256, 0, stream>>>(res_W1, wRes1,  12288);
  cvt_kernel<<<48, 256, 0, stream>>>(res_W2, wRes2,  12288);
  cvt_kernel<<<16, 256, 0, stream>>>(W_last, wLast16, 4096);

  long n4 = NF / 4;
  zero_kernel<<<(int)((n4 + 255) / 256), 256, 0, stream>>>(aggr, n4);

  node_linear_kernel<<<(N + 127) / 128, 256, 32768, stream>>>(
      x, wDiff16, b_diff, wSame16, b_same, hbuf, outMsg, N);
  edge_kernel<<<E / 128, 256, 0, stream>>>(edge_attr, edge_index, wG16, hbuf, aggr, E);
  residual_kernel<<<(N + 63) / 64, 256, 32768, stream>>>(
      x, aggr, wRes1, res_b1, wRes2, res_b2, wLast16, b_last, u, out0, outMsg, N);
}
